// TripletLRSelfSupModel_3264175145316
// MI455X (gfx1250) — matrix operand order repaired and checked by reading, not compiled, not run on hardware
//
#include <hip/hip_runtime.h>
#include <hip/hip_bf16.h>

// ---------------------------------------------------------------------------
// MI455X (gfx1250, wave32) implementation.
//   conv1 (3->32,s1): direct VALU conv, weights staged in LDS, NHWC f16 out.
//   conv2/conv3 (s2): implicit-GEMM on v_wmma_f32_16x16x32_f16, one 16x16
//     output tile per wave; boundary handling is branch-free (clamp+select)
//     so the K loop has no exec-mask churn; global_prefetch for next tap.
//   correlation: per 16-pixel row tile, per oy: two 16x16x128 WMMA chains of
//     pairwise dots a[16] x b[32]; b-rows double-buffered into LDS with the
//     CDNA5 async global->LDS engine (s_wait_asynccnt) when available;
//     diagonal band extracted through LDS; softmax over 169 offsets.
//   loss: elementwise + LDS tree reduction + global f32 atomics.
// ---------------------------------------------------------------------------

typedef __attribute__((ext_vector_type(16))) _Float16 v16h;
typedef __attribute__((ext_vector_type(8)))  _Float16 v8h;
typedef __attribute__((ext_vector_type(8)))  float    v8f;

union V16 { v16h v; v8h h[2]; };

#if defined(__HIP_DEVICE_COMPILE__) && \
    __has_builtin(__builtin_amdgcn_global_load_async_to_lds_b128) && \
    __has_builtin(__builtin_amdgcn_s_wait_asynccnt)
#define CORR_ASYNC 1
// Parameter type per hipcc diagnostic: int __attribute__((vector_size(16)))*
typedef int v4i_b128 __attribute__((vector_size(16)));
#else
#define CORR_ASYNC 0
#endif

static __device__ inline v8f wmma_f16(v16h a, v16h b, v8f c) {
  // D = A(16x32 f16) * B(32x16 f16) + C(16x16 f32)
  return __builtin_amdgcn_wmma_f32_16x16x32_f16(
      /*neg_a=*/false, a, /*neg_b=*/false, b,
      /*c_mod=*/(short)0, c, /*reuse_a=*/false, /*reuse_b=*/false);
}

static __device__ inline int clampi(int v, int lo, int hi) {
  return v < lo ? lo : (v > hi ? hi : v);
}

// ---------------------------------------------------------------------------
// Kernel 1: conv1 3->32, stride 1, pad 1, 512x512, ReLU, NHWC f16 output.
// ---------------------------------------------------------------------------
__global__ __launch_bounds__(256)
void conv1_kernel(const float* __restrict__ img, const float* __restrict__ w1,
                  const float* __restrict__ b1, _Float16* __restrict__ out) {
  __shared__ float ws[288];
  __shared__ float bs[32];
  const int tid = threadIdx.x;
  for (int i = tid; i < 288; i += 256) ws[i] = w1[i];
  if (tid < 32) bs[tid] = b1[tid];
  __syncthreads();

  const int idx = blockIdx.x * 256 + tid;        // 4*512*512 threads
  const int b = idx >> 18;
  const int rem = idx & 262143;
  const int y = rem >> 9;
  const int x = rem & 511;

  float acc[32];
#pragma unroll
  for (int oc = 0; oc < 32; ++oc) acc[oc] = bs[oc];

  for (int ic = 0; ic < 3; ++ic) {
#pragma unroll
    for (int t = 0; t < 9; ++t) {
      const int ky = t / 3, kx = t % 3;
      const int iy = y + ky - 1, ix = x + kx - 1;
      const bool ok = (iy >= 0) && (iy < 512) && (ix >= 0) && (ix < 512);
      const int iyc = clampi(iy, 0, 511), ixc = clampi(ix, 0, 511);
      float v = img[(((b * 3 + ic) * 512 + iyc) * 512) + ixc];
      v = ok ? v : 0.0f;
#pragma unroll
      for (int oc = 0; oc < 32; ++oc)
        acc[oc] += v * ws[(oc * 3 + ic) * 9 + t];
    }
  }

  _Float16* o = out + (size_t)idx * 32;
#pragma unroll
  for (int oc = 0; oc < 32; ++oc) {
    float r = acc[oc];
    o[oc] = (_Float16)(r > 0.0f ? r : 0.0f);
  }
}

// ---------------------------------------------------------------------------
// Weight repack: OIHW f32 -> [ky][kx][co][ci] f16 (ci contiguous -> B-fragment
// loads are two contiguous v8h per lane).
// ---------------------------------------------------------------------------
__global__ void prep_w_kernel(const float* __restrict__ w, _Float16* __restrict__ o,
                              int O, int I, int n) {
  const int idx = blockIdx.x * 256 + threadIdx.x;
  if (idx >= n) return;
  const int oc = idx / (I * 9);
  const int r  = idx % (I * 9);
  const int ic = r / 9;
  const int t  = r % 9;
  o[((size_t)t * O + oc) * I + ic] = (_Float16)w[idx];
}

// ---------------------------------------------------------------------------
// Kernel 2: stride-2 3x3 conv as implicit GEMM on WMMA.
//   wave tile: M=16 output pixels (one row segment) x N=16 output channels.
//   K loop: 9 taps x (Cin/32) chunks of v_wmma_f32_16x16x32_f16.
//   Boundary handling: clamp address + v_cndmask zeroing (no exec branches).
// in: NHWC f16 [b][Hin][Hin][Cin], wgt: [tap][co][ci] f16, out NHWC f16.
// ---------------------------------------------------------------------------
__global__ __launch_bounds__(256)
void conv_s2_wmma_kernel(const _Float16* __restrict__ in,
                         const _Float16* __restrict__ wgt,
                         const float* __restrict__ bias,
                         _Float16* __restrict__ out,
                         int Cin, int Cout, int Hin, int Hout) {
  const int lane = threadIdx.x & 31;
  int wave = blockIdx.x * (blockDim.x >> 5) + (threadIdx.x >> 5);
  const int ntN = Cout >> 4, tilesX = Hout >> 4;
  const int nt = wave % ntN; wave /= ntN;
  const int xt = wave % tilesX; wave /= tilesX;
  const int yo = wave % Hout;
  const int b  = wave / Hout;

  const int m  = lane & 15;   // A row / output pixel within tile
  const int hi = lane >> 4;   // lane-group selector in fragment layouts
  const int xo_a = xt * 16 + m;

  const v8h zero8 = {};
  v8f c = {};

  for (int tap = 0; tap < 9; ++tap) {
    const int ky = tap / 3, kx = tap % 3;
    const int iy = 2 * yo + ky - 1;
    const int ix = 2 * xo_a + kx - 1;
    const bool ok = (iy >= 0) && (iy < Hin) && (ix >= 0) && (ix < Hin);
    const int iyc = clampi(iy, 0, Hin - 1);
    const int ixc = clampi(ix, 0, Hin - 1);
    const _Float16* ab = in + ((size_t)((b * Hin + iyc) * Hin + ixc)) * Cin;

    // Prefetch next tap's input row segment (global_prefetch_b8).
    if (tap < 8) {
      const int ky2 = (tap + 1) / 3, kx2 = (tap + 1) % 3;
      const int iy2 = clampi(2 * yo + ky2 - 1, 0, Hin - 1);
      const int ix2 = clampi(2 * xo_a + kx2 - 1, 0, Hin - 1);
      __builtin_prefetch(in + ((size_t)((b * Hin + iy2) * Hin + ix2)) * Cin, 0, 0);
    }

    for (int cb = 0; cb < Cin; cb += 32) {
      // A fragment (16x32 f16): lane row m; halves 0..7 -> K = hi*8+0..7,
      // halves 8..15 -> K = 16+hi*8+0..7  (CDNA5 16-bit A layout).
      V16 a;
      a.h[0] = *(const v8h*)(ab + cb + hi * 8);
      a.h[1] = *(const v8h*)(ab + cb + 16 + hi * 8);
      a.h[0] = ok ? a.h[0] : zero8;
      a.h[1] = ok ? a.h[1] : zero8;
      // B fragment (32x16 f16): lane column n = lane&15, K same as A layout.
      V16 bb;
      const _Float16* wb =
          wgt + ((size_t)(tap * Cout + nt * 16 + (lane & 15))) * Cin + cb + hi * 8;
      bb.h[0] = *(const v8h*)(wb);
      bb.h[1] = *(const v8h*)(wb + 16);
      c = wmma_f16(a.v, bb.v, c);
    }
  }

  // C fragment: lane L, vgpr v -> (M = v + hi*8, N = L&15)
  const int co = nt * 16 + (lane & 15);
  const float bv = bias[co];
#pragma unroll
  for (int v = 0; v < 8; ++v) {
    const int xo = xt * 16 + v + hi * 8;
    float r = c[v] + bv;
    r = r > 0.0f ? r : 0.0f;
    out[((size_t)((b * Hout + yo) * Hout + xo)) * Cout + co] = (_Float16)r;
  }
}

// ---------------------------------------------------------------------------
// Kernel 3: in-place channel-L2 normalization, NHWC f16 [b][128][128][128].
// ---------------------------------------------------------------------------
__global__ __launch_bounds__(256)
void l2norm_kernel(_Float16* __restrict__ f) {
  const int idx = blockIdx.x * 256 + threadIdx.x;   // 65536 pixels
  _Float16* p = f + (size_t)idx * 128;
  float s = 0.0f;
#pragma unroll
  for (int k = 0; k < 16; ++k) {
    v8h h = *(const v8h*)(p + k * 8);
#pragma unroll
    for (int e = 0; e < 8; ++e) { float v = (float)h[e]; s += v * v; }
  }
  const float scale = 1.0f / fmaxf(sqrtf(s), 1e-12f);
#pragma unroll
  for (int k = 0; k < 16; ++k) {
    v8h h = *(const v8h*)(p + k * 8);
#pragma unroll
    for (int e = 0; e < 8; ++e) h[e] = (_Float16)((float)h[e] * scale);
    *(v8h*)(p + k * 8) = h;
  }
}

// ---------------------------------------------------------------------------
// Kernel 4: local correlation (R=6 -> 13x13=169) + softmax + expected offsets.
// One wave per (b, y, 16-pixel x-tile). For each oy, two WMMA chains
// (K=128 in 4 chunks) compute all pairwise dots a[16] x b[32]; the needed
// 13-wide diagonal band goes to LDS, then softmax over 169 per pixel.
// The b-row (32 pixels x 128ch = 8KB) for oy+1 is double-buffered into LDS
// with async global->LDS copies overlapping the current oy's WMMAs.
// ---------------------------------------------------------------------------
__global__ __launch_bounds__(32)
void corr_kernel(const _Float16* __restrict__ fa, const _Float16* __restrict__ fb,
                 float* __restrict__ mux, float* __restrict__ muy) {
  __shared__ float corr_s[16][176];   // [pixel][169 offsets, padded]
  __shared__ float cmat[32][16];      // [b-col j][a-row m]
#if CORR_ASYNC
  __shared__ _Float16 brow[2][32][128];  // double-buffered b-row tiles
#endif

  int id = blockIdx.x;                // 4 * 128 * 8
  const int xt = id & 7;  id >>= 3;
  const int y  = id & 127;
  const int b  = id >> 7;
  const int x0 = xt * 16;

  const int lane = threadIdx.x;
  const int m  = lane & 15;
  const int hi = lane >> 4;
  const v8h zero8 = {};

  // Preload the 4 A-fragment chunks (pixel (y, x0+m), 128 channels).
  V16 aF[4];
  {
    const _Float16* ab = fa + ((size_t)((b * 128 + y) * 128 + (x0 + m))) * 128;
#pragma unroll
    for (int cch = 0; cch < 4; ++cch) {
      aF[cch].h[0] = *(const v8h*)(ab + 32 * cch + hi * 8);
      aF[cch].h[1] = *(const v8h*)(ab + 32 * cch + 16 + hi * 8);
    }
  }

#if CORR_ASYNC
  // Issue one b-row (32 pixels x 256B) as 16 async 16B/lane transfers.
  auto issue_row = [&](int oy, int buf) {
    const int ybc = clampi(y + oy - 6, 0, 127);
#pragma unroll
    for (int i = 0; i < 16; ++i) {
      const int e = i * 32 + lane;   // 0..511
      const int p = e >> 4;          // pixel 0..31
      const int q = e & 15;          // 16B chunk (8 halfs) within pixel
      const int xbc = clampi(x0 - 6 + p, 0, 127);
      const _Float16* g =
          fb + ((size_t)((b * 128 + ybc) * 128 + xbc)) * 128 + q * 8;
      __builtin_amdgcn_global_load_async_to_lds_b128(
          (v4i_b128*)g, (v4i_b128*)&brow[buf][p][q * 8], 0, 0);
    }
  };
  issue_row(0, 0);
#endif

  for (int oy = 0; oy < 13; ++oy) {
    const int yb = y + oy - 6;
    const int ybc = clampi(yb, 0, 127);
#if CORR_ASYNC
    if (oy < 12) {
      issue_row(oy + 1, (oy + 1) & 1);
      __builtin_amdgcn_s_wait_asynccnt(16);  // current row resident
    } else {
      __builtin_amdgcn_s_wait_asynccnt(0);
    }
#endif
#pragma unroll
    for (int g2 = 0; g2 < 2; ++g2) {
      const int j  = (lane & 15) + g2 * 16;   // B column = b-pixel x0-6+j
      const int xb = x0 - 6 + j;
      const int xbc = clampi(xb, 0, 127);
      const bool ok = (yb >= 0) && (yb < 128) && (xb >= 0) && (xb < 128);
      v8f cf = {};
#pragma unroll
      for (int cch = 0; cch < 4; ++cch) {
        V16 bF;
#if CORR_ASYNC
        const _Float16* lb = &brow[oy & 1][j][cch * 32 + hi * 8];
        bF.h[0] = *(const v8h*)(lb);
        bF.h[1] = *(const v8h*)(lb + 16);
#else
        const _Float16* bbp =
            fb + ((size_t)((b * 128 + ybc) * 128 + xbc)) * 128 + 32 * cch + hi * 8;
        bF.h[0] = *(const v8h*)(bbp);
        bF.h[1] = *(const v8h*)(bbp + 16);
#endif
        bF.h[0] = ok ? bF.h[0] : zero8;
        bF.h[1] = ok ? bF.h[1] : zero8;
        cf = wmma_f16(aF[cch].v, bF.v, cf);
      }
#pragma unroll
      for (int v = 0; v < 8; ++v)
        cmat[(lane & 15) + g2 * 16][v + hi * 8] = cf[v];
    }
    __syncthreads();
    // Extract diagonal band: corr(pixel mm, ox) = dot(a[mm], b[x0-6+mm+ox]).
    for (int n = lane; n < 208; n += 32) {
      const int ox = n >> 4;
      const int mm = n & 15;
      corr_s[mm][oy * 13 + ox] = cmat[mm + ox][mm] * (1.0f / 0.07f);
    }
    __syncthreads();
  }

  // Softmax over 169 offsets + expected dx/dy (lanes 0..15, one pixel each).
  if (lane < 16) {
    float mx = -3.4e38f;
    for (int n = 0; n < 169; ++n) mx = fmaxf(mx, corr_s[lane][n]);
    float s = 0.0f, sx = 0.0f, sy = 0.0f;
    for (int n = 0; n < 169; ++n) {
      const float e = __expf(corr_s[lane][n] - mx);
      s  += e;
      sx += e * (float)((n % 13) - 6);
      sy += e * (float)((n / 13) - 6);
    }
    const int out_i = (b * 128 + y) * 128 + x0 + lane;
    mux[out_i] = sx / s;
    muy[out_i] = sy / s;
  }
}

// ---------------------------------------------------------------------------
// Kernel 5: loss terms + reduction.
// ---------------------------------------------------------------------------
__global__ void init_acc_kernel(float* acc) {
  if (threadIdx.x < 2) acc[threadIdx.x] = 0.0f;
}

__global__ __launch_bounds__(256)
void loss_kernel(const float* __restrict__ mu1x, const float* __restrict__ mu2x,
                 const float* __restrict__ mu2y, const float* __restrict__ tau1,
                 const float* __restrict__ tau2, const float* __restrict__ Kmat,
                 float* __restrict__ acc) {
  const int tid = threadIdx.x;
  const int idx = blockIdx.x * 256 + tid;       // 4*128*128
  const int b  = idx >> 14;
  const int rem = idx & 16383;
  const int xx = rem & 127;

  const float m1x = mu1x[idx], m2x = mu2x[idx], m2y = mu2y[idx];
  const float t1 = tau1[b], t2 = tau2[b];
  const float fx = Kmat[b * 9];
  const float sg = (t1 > 0.0f) ? 1.0f : ((t1 < 0.0f) ? -1.0f : 0.0f);

  const float depth_raw = fx * t1 / (m1x * 4.0f + 0.001f * sg);
  const float qpx = (float)xx + m1x * (t2 / (t1 + 0.001f * sg));
  const float qax = (float)xx + m2x;
  const float dyv = -m2y;                       // q_pred_y - q_actual_y
  const bool finite = (depth_raw * 0.0f == 0.0f);   // false for NaN/Inf
  const bool valid = finite && (depth_raw > 0.0f) &&
                     (qpx >= 0.0f) && (qpx <= 127.0f);
  const float vf = valid ? 1.0f : 0.0f;
  const float pl = (qpx - qax) * (qpx - qax) + dyv * dyv;

  __shared__ float s1[256];
  __shared__ float s2[256];
  s1[tid] = pl * vf;
  s2[tid] = vf;
  __syncthreads();
  for (int off = 128; off > 0; off >>= 1) {
    if (tid < off) { s1[tid] += s1[tid + off]; s2[tid] += s2[tid + off]; }
    __syncthreads();
  }
  if (tid == 0) {
    atomicAdd(&acc[0], s1[0]);
    atomicAdd(&acc[1], s2[0]);
  }
}

__global__ void finalize_kernel(const float* __restrict__ acc, float* __restrict__ out) {
  out[0] = acc[0] / fmaxf(acc[1], 1.0f);
}

// ---------------------------------------------------------------------------
// Host launcher.
// ---------------------------------------------------------------------------
extern "C" void kernel_launch(void* const* d_in, const int* in_sizes, int n_in,
                              void* d_out, int out_size, void* d_ws, size_t ws_size,
                              hipStream_t stream) {
  (void)in_sizes; (void)n_in; (void)out_size; (void)ws_size;
  const float* img[3] = { (const float*)d_in[0], (const float*)d_in[1],
                          (const float*)d_in[2] };
  const float* tau1 = (const float*)d_in[3];
  const float* tau2 = (const float*)d_in[4];
  const float* Kmat = (const float*)d_in[5];
  const float* w1 = (const float*)d_in[6];
  const float* b1 = (const float*)d_in[7];
  const float* w2 = (const float*)d_in[8];
  const float* b2 = (const float*)d_in[9];
  const float* w3 = (const float*)d_in[10];
  const float* b3 = (const float*)d_in[11];

  char* ws = (char*)d_ws;
  size_t off = 0;
  auto take = [&](size_t bytes) {
    char* p = ws + off;
    off += (bytes + 255) & ~(size_t)255;
    return p;
  };
  _Float16* W2H  = (_Float16*)take((size_t)9 * 64 * 32 * 2);          // 36 KB
  _Float16* W3H  = (_Float16*)take((size_t)9 * 128 * 64 * 2);         // 144 KB
  _Float16* TMP1 = (_Float16*)take((size_t)4 * 512 * 512 * 32 * 2);   // 64 MB
  _Float16* TMP2 = (_Float16*)take((size_t)4 * 256 * 256 * 64 * 2);   // 32 MB
  _Float16* F3[3];
  for (int i = 0; i < 3; ++i)
    F3[i] = (_Float16*)take((size_t)4 * 128 * 128 * 128 * 2);         // 16 MB ea
  float* MU1X = (float*)take((size_t)4 * 128 * 128 * 4);
  float* MU2X = (float*)take((size_t)4 * 128 * 128 * 4);
  float* MU2Y = (float*)take((size_t)4 * 128 * 128 * 4);
  float* ACC  = (float*)take(256);

  // Weight repacks (f32 OIHW -> f16 [tap][co][ci]).
  prep_w_kernel<<<72, 256, 0, stream>>>(w2, W2H, 64, 32, 18432);
  prep_w_kernel<<<288, 256, 0, stream>>>(w3, W3H, 128, 64, 73728);

  // Encoder for the three images (buffers reused per image).
  for (int i = 0; i < 3; ++i) {
    conv1_kernel<<<4096, 256, 0, stream>>>(img[i], w1, b1, TMP1);
    conv_s2_wmma_kernel<<<8192, 256, 0, stream>>>(TMP1, W2H, b2, TMP2,
                                                  32, 64, 512, 256);
    conv_s2_wmma_kernel<<<4096, 256, 0, stream>>>(TMP2, W3H, b3, F3[i],
                                                  64, 128, 256, 128);
  }

  // L2 normalization (in place).
  for (int i = 0; i < 3; ++i)
    l2norm_kernel<<<256, 256, 0, stream>>>(F3[i]);

  // Correlation + softmax + expected offsets.
  corr_kernel<<<4096, 32, 0, stream>>>(F3[0], F3[1], MU1X, MU2Y);  // MU2Y: scratch
  corr_kernel<<<4096, 32, 0, stream>>>(F3[1], F3[2], MU2X, MU2Y);

  // Loss reduction.
  init_acc_kernel<<<1, 32, 0, stream>>>(ACC);
  loss_kernel<<<256, 256, 0, stream>>>(MU1X, MU2X, MU2Y, tau1, tau2, Kmat, ACC);
  finalize_kernel<<<1, 1, 0, stream>>>(ACC, (float*)d_out);
}
